// PhaseFieldPredictor_7404523619171
// MI455X (gfx1250) — matrix operand order repaired and checked by reading, not compiled, not run on hardware
//
#include <hip/hip_runtime.h>

#define NB 16
#define NTOK 10
#define ND 32
#define NLIFT 128
#define NHEADS 2
#define NDH 16
#define NM1 8
#define NM2 8
#define NLAY 4
#define HH 64
#define WW 64
#define HP 70
#define WP 70
#define PPIX (HP*WP)          // 4900
#define NMODE 128             // 16 x 8
#define NBT (NB*NTOK)         // 160
#define PI_F 3.14159265358979f

typedef _Float16 v16h __attribute__((ext_vector_type(16)));
typedef _Float16 v8h  __attribute__((ext_vector_type(8)));
typedef float v8f __attribute__((ext_vector_type(8)));

__device__ __forceinline__ float gelu_f(float x){
    float x3 = x*x*x;
    return 0.5f*x*(1.f + tanhf(0.7978845608f*(x + 0.044715f*x3)));
}

// ---- WMMA operand assembly (wave32) -----------------------------------------
// A (16x32 f16, MxK): lane L: M=L%16; per-lane halves are one contiguous run
// of 16: K in [kb,kb+16), kb = (L<16)?0:16  ->  2x ds_load_b128.
__device__ __forceinline__ v16h load_a_lds(const _Float16* base, int ld){
    int lane = threadIdx.x & 31;
    int m = lane & 15;
    int kb = (lane < 16) ? 0 : 16;
    v8h lo = *(const v8h*)(base + m*ld + kb);
    v8h hi = *(const v8h*)(base + m*ld + kb + 8);
    return __builtin_shufflevector(lo, hi, 0,1,2,3,4,5,6,7,8,9,10,11,12,13,14,15);
}
// B fragments are PRE-PACKED lane-major: lane L's 16 halves contiguous.
__device__ __forceinline__ v16h load_b_packed(const _Float16* base){
    int lane = threadIdx.x & 31;
    return *(const v16h*)(base + lane*16);
}
__device__ __forceinline__ v8f wmma16(v16h a, v16h b, v8f c){
    return __builtin_amdgcn_wmma_f32_16x16x32_f16(false, a, false, b, (short)0, c, false, false);
}

// ---- CDNA5 async global->LDS copy (ASYNCcnt) --------------------------------
__device__ __forceinline__ void async_copy_b128(void* lds_dst, const void* gsrc){
    unsigned l = (unsigned)(uintptr_t)lds_dst;     // low 32 bits = LDS offset
    asm volatile("global_load_async_to_lds_b128 %0, %1, off"
                 :: "v"(l), "v"(gsrc) : "memory");
}
__device__ __forceinline__ void wait_async(){
    asm volatile("s_wait_asynccnt 0x0" ::: "memory");
}

// ---- utility kernels --------------------------------------------------------
__global__ void k_zero(float* p, long n){
    long i = (long)blockIdx.x*blockDim.x + threadIdx.x;
    long s = (long)gridDim.x*blockDim.x;
    for (; i < n; i += s) p[i] = 0.f;
}
__global__ void k_add(float* z, const float* o, long n){
    long i = (long)blockIdx.x*blockDim.x + threadIdx.x;
    long s = (long)gridDim.x*blockDim.x;
    for (; i < n; i += s) z[i] += o[i];
}

// ---- weight prep: all B operands packed lane-major per 32x16 fragment -------
// lift_w2 [32][128] -> frag pack [kt(4)][nt(2)][lane(32)][j(16)]
__global__ void k_pack_liftw2(const float* __restrict__ w2, _Float16* __restrict__ dst){
    int i = blockIdx.x*blockDim.x + threadIdx.x;
    if (i >= NLIFT*ND) return;
    int j = i & 15, lane = (i >> 4) & 31, nt = (i >> 9) & 1, kt = i >> 10;
    int k = kt*32 + ((lane < 16) ? 0 : 16) + j;      // f index
    int n = nt*16 + (lane & 15);                     // d index
    dst[i] = (_Float16)w2[n*NLIFT + k];
}
// forward-x DFT table: logical [96][16] -> frag pack [kt(3)][lane][j]
__global__ void k_pack_ftab(_Float16* __restrict__ dst){
    int i = blockIdx.x*blockDim.x + threadIdx.x;
    if (i >= 3*512) return;
    int j = i & 15, lane = (i >> 4) & 31, kt = i >> 9;
    int xr = kt*32 + ((lane < 16) ? 0 : 16) + j;
    int n = lane & 15;
    float v = 0.f;
    if (xr < WP){
        float ang = -2.f*PI_F*(float)(n & 7)*(float)xr/(float)WP;
        v = (n < 8) ? cosf(ang) : sinf(ang);
    }
    dst[i] = (_Float16)v;
}
// spectral weights [2][32][32][8][8][2] -> per-mode frag pack
// [mode(128)][kt(2)][nt(4)][lane(32)][j(16)]  (real-packed complex GEMM)
__global__ void k_pack_spec(const float* __restrict__ w, _Float16* __restrict__ dst){
    int i = blockIdx.x*blockDim.x + threadIdx.x;
    if (i >= NMODE*4096) return;
    int j = i & 15, lane = (i >> 4) & 31, nt = (i >> 9) & 3, kt = (i >> 11) & 1, m = i >> 12;
    int k = kt*32 + ((lane < 16) ? 0 : 16) + j;
    int n = nt*16 + (lane & 15);
    int kh = m >> 3, kw = m & 7;
    int corner = (kh >= 8) ? 1 : 0; int khh = kh & 7;
    int ci = k & 31, o = n & 31;
    const float* base = w + ((((corner*ND + ci)*ND + o)*NM1 + khh)*NM2 + kw)*2;
    float re = base[0], im = base[1];
    float v = (k < 32) ? ((n < 32) ? re : im) : ((n < 32) ? -im : re);
    dst[i] = (_Float16)v;
}

// ---- fusion 3x3 conv (50 -> 10 ch) -----------------------------------------
__global__ void k_fusion(const float* __restrict__ x, const float* __restrict__ w,
                         const float* __restrict__ bias, float* __restrict__ fused){
    __shared__ float sw[4500];
    for (int i = threadIdx.x; i < 4500; i += blockDim.x) sw[i] = w[i];
    __syncthreads();
    int idx = blockIdx.x*blockDim.x + threadIdx.x;
    if (idx >= NB*NTOK*HH*WW) return;
    int xp = idx & 63, yp = (idx >> 6) & 63;
    int co = (idx >> 12) % NTOK; int b = idx / (NTOK*HH*WW);
    float acc = bias[co];
    const float* xb = x + (long)b*50*HH*WW;
    const float* wc = sw + co*450;
    for (int ci = 0; ci < 50; ++ci){
        const float* xc = xb + ci*HH*WW;
        const float* wk = wc + ci*9;
#pragma unroll
        for (int ky = 0; ky < 3; ++ky){
            int yy = yp + ky - 1; if ((unsigned)yy >= HH) continue;
#pragma unroll
            for (int kx = 0; kx < 3; ++kx){
                int xx = xp + kx - 1; if ((unsigned)xx >= WW) continue;
                acc += xc[yy*WW + xx]*wk[ky*3 + kx];
            }
        }
    }
    fused[idx] = acc;
}

// ---- lift MLP 1->128->32 via WMMA, write into zero-padded z -----------------
__global__ void k_lift(const float* __restrict__ fused, const float* __restrict__ w1,
                       const float* __restrict__ b1, const _Float16* __restrict__ w2h,
                       const float* __restrict__ b2, float* __restrict__ z){
    __shared__ __align__(32) _Float16 sA[16][NLIFT];
    int lane = threadIdx.x & 31;
    int tile = blockIdx.x;                    // 160*256 tiles of 16 pixels
    int bt = tile >> 8, pg = tile & 255;
    const float* fp = fused + (long)bt*HH*WW + pg*16;
    {
        int p = lane & 15; int f0 = (lane < 16) ? 0 : 64;
        float v = fp[p];
        for (int j = 0; j < 64; ++j){
            int f = f0 + j;
            sA[p][f] = (_Float16)gelu_f(w1[f]*v + b1[f]);
        }
    }
    __syncthreads();
    __builtin_prefetch((const void*)w2h, 0, 1);
    for (int nt = 0; nt < 2; ++nt){
        v8f acc = {};
#pragma unroll
        for (int kt = 0; kt < 4; ++kt){
            v16h a = load_a_lds(&sA[0][0] + kt*32, NLIFT);
            v16h b = load_b_packed(w2h + (kt*2 + nt)*512);
            acc = wmma16(a, b, acc);
        }
        int N = lane & 15; int o = nt*16 + N;
        float bb = b2[o];
#pragma unroll
        for (int r = 0; r < 8; ++r){
            int M = 8*(lane >> 4) + r;
            int px = pg*16 + M;
            int y = px >> 6, xx = px & 63;
            z[((long)bt*ND + o)*PPIX + y*WP + xx] = acc[r] + bb;
        }
    }
}

// ---- forward truncated DFT along x: rows [BTD*70 x 70] @ [70x(8re+8im)] -----
__global__ void k_dft_x(const float* __restrict__ z, const _Float16* __restrict__ ftab,
                        float* __restrict__ P){
    __shared__ __align__(32) _Float16 sA[16][96];
    int lane = threadIdx.x & 31;
    long row0 = (long)blockIdx.x*16;          // rows = btd*70 + y (358400 total)
    {
        int r = lane & 15; int c0 = (lane < 16) ? 0 : 48;
        const float* zr = z + (row0 + r)*WP;
        for (int j = 0; j < 48; ++j){
            int c = c0 + j;
            sA[r][c] = (c < WP) ? (_Float16)zr[c] : (_Float16)0.f;
        }
    }
    __syncthreads();
    v8f acc = {};
#pragma unroll
    for (int kt = 0; kt < 3; ++kt){
        v16h a = load_a_lds(&sA[0][0] + kt*32, 96);
        v16h b = load_b_packed(ftab + kt*512);
        acc = wmma16(a, b, acc);
    }
    int N = lane & 15; int kw = N & 7, part = N >> 3;
#pragma unroll
    for (int r = 0; r < 8; ++r){
        long row = row0 + 8*(lane >> 4) + r;
        P[row*16 + kw*2 + part] = acc[r];
    }
}

// ---- forward DFT along y -> mode-major Zm[m][bt][re/im][ch] -----------------
__global__ void k_dft_y(const float* __restrict__ P, float* __restrict__ Zm){
    int i = blockIdx.x*blockDim.x + threadIdx.x;   // btd*128 + kh*8+kw
    if (i >= NBT*ND*NMODE) return;
    int m = i & 127; long btd = i >> 7;
    int bt = (int)(btd >> 5), d = (int)(btd & 31);
    int kh = m >> 3, kw = m & 7;
    float khf = (kh < 8) ? (float)kh : (float)(kh - 16);
    const float* Pr = P + btd*HP*16 + kw*2;
    float ar = 0.f, ai = 0.f;
    for (int y = 0; y < HP; ++y){
        float s, c; __sincosf(-2.f*PI_F*khf*(float)y/(float)HP, &s, &c);
        float pre = Pr[y*16], pim = Pr[y*16 + 1];
        ar += c*pre - s*pim;
        ai += c*pim + s*pre;
    }
    long o = ((long)m*NBT + bt)*64 + d;
    Zm[o] = ar; Zm[o + 32] = ai;
}

// ---- per-mode complex channel mix: [160 x 64] @ [64 x 64] real-packed -------
// In/Out mode-major: row (m,bt) = 64 contiguous floats [re ch0..31 | im ch0..31]
__global__ void k_modemix(const float* __restrict__ In, const _Float16* __restrict__ Wp,
                          float* __restrict__ Out){
    __shared__ __align__(32) _Float16 sA[16][64];
    __shared__ __align__(32) _Float16 sB[4096];
    int lane = threadIdx.x & 31;
    int mode = blockIdx.x;                    // 128
    int mt = blockIdx.y;                      // 10 tiles of 16 (b,t) rows
    const _Float16* Wm = Wp + (long)mode*4096;
    // async-stage the mode's packed B fragments (8KB) into LDS
#pragma unroll
    for (int it = 0; it < 16; ++it)
        async_copy_b128(&sB[it*256 + lane*8], Wm + it*256 + lane*8);
    // stage A (coalesced: each row is 64 contiguous floats)
    {
        int r = lane & 15; int k0 = (lane < 16) ? 0 : 32;
        const float4* s4 = (const float4*)(In + ((long)mode*NBT + mt*16 + r)*64 + k0);
#pragma unroll
        for (int q = 0; q < 4; ++q){
            float4 a0 = s4[2*q], a1 = s4[2*q + 1];
            v8h hv;
            hv[0] = (_Float16)a0.x; hv[1] = (_Float16)a0.y;
            hv[2] = (_Float16)a0.z; hv[3] = (_Float16)a0.w;
            hv[4] = (_Float16)a1.x; hv[5] = (_Float16)a1.y;
            hv[6] = (_Float16)a1.z; hv[7] = (_Float16)a1.w;
            *(v8h*)&sA[r][k0 + 8*q] = hv;
        }
    }
    wait_async();
    __syncthreads();
    for (int nt = 0; nt < 4; ++nt){
        v8f acc = {};
#pragma unroll
        for (int kt = 0; kt < 2; ++kt){
            v16h a = load_a_lds(&sA[0][0] + kt*32, 64);
            v16h b = load_b_packed(sB + (kt*4 + nt)*512);
            acc = wmma16(a, b, acc);
        }
        int N = lane & 15; int col = nt*16 + N;
        int o = col & 31, part = col >> 5;
#pragma unroll
        for (int r = 0; r < 8; ++r){
            int bt = mt*16 + 8*(lane >> 4) + r;
            Out[((long)mode*NBT + bt)*64 + part*32 + o] = acc[r];
        }
    }
}

// ---- attention scores via Parseval over the 128 retained modes --------------
__global__ void k_scores(const float* __restrict__ Qm, const float* __restrict__ Km,
                         float* __restrict__ scores){
    __shared__ float red[256];
    int blk = blockIdx.x;                     // b*200 + h*100 + i*10 + j
    int j = blk % 10, i2 = (blk/10) % 10, h = (blk/100) % 2, b = blk/200;
    float acc = 0.f;
    for (int n = threadIdx.x; n < NDH*NMODE; n += 256){
        int d = h*NDH + (n >> 7); int m = n & 127;
        float mult = ((m & 7) == 0) ? 1.f : 2.f;     // Hermitian mirror count
        long qi = ((long)m*NBT + b*NTOK + i2)*64 + d;
        long ki = ((long)m*NBT + b*NTOK + j )*64 + d;
        acc += mult*(Qm[qi]*Km[ki] + Qm[qi + 32]*Km[ki + 32]);
    }
    red[threadIdx.x] = acc; __syncthreads();
    for (int s = 128; s > 0; s >>= 1){
        if (threadIdx.x < s) red[threadIdx.x] += red[threadIdx.x + s];
        __syncthreads();
    }
    if (threadIdx.x == 0)
        scores[blk] = red[0]*(1.f/(float)(HP*WP))*rsqrtf((float)(NDH*HP*WP));
}
__global__ void k_softmax(float* scores){
    int r = threadIdx.x;                      // (b,h,i): 320 rows of 10
    if (r >= NB*NHEADS*NTOK) return;
    float* s = scores + r*NTOK;
    float mx = s[0];
    for (int j = 1; j < NTOK; ++j) mx = fmaxf(mx, s[j]);
    float e[NTOK]; float sum = 0.f;
    for (int j = 0; j < NTOK; ++j){ e[j] = __expf(s[j] - mx); sum += e[j]; }
    float inv = 1.f/sum;
    for (int j = 0; j < NTOK; ++j) s[j] = e[j]*inv;
}
__global__ void k_attn_apply(const float* __restrict__ attn, const float* __restrict__ Vm,
                             float* __restrict__ Om){
    long i = (long)blockIdx.x*blockDim.x + threadIdx.x;  // (b,i,d,m)
    if (i >= (long)NBT*ND*NMODE) return;
    int m = (int)(i & 127); long t = i >> 7; int d = (int)(t & 31); t >>= 5;
    int tok = (int)(t % NTOK); int b = (int)(t/NTOK);
    int h = d >> 4;
    const float* a = attn + ((b*NHEADS + h)*NTOK + tok)*NTOK;
    float ar = 0.f, ai = 0.f;
    for (int j = 0; j < NTOK; ++j){
        long vb = ((long)m*NBT + b*NTOK + j)*64 + d;
        ar += a[j]*Vm[vb]; ai += a[j]*Vm[vb + 32];
    }
    long ob = ((long)m*NBT + b*NTOK + tok)*64 + d;
    Om[ob] = ar; Om[ob + 32] = ai;
}

// ---- inverse DFT along y (16 modes -> 70) -----------------------------------
__global__ void k_idft_y(const float* __restrict__ Mm, float* __restrict__ T1){
    long i = (long)blockIdx.x*blockDim.x + threadIdx.x;  // btd*70*8 + y*8 + kw
    if (i >= (long)NBT*ND*HP*NM2) return;
    int kw = (int)(i & 7); long t = i >> 3; int y = (int)(t % HP); long btd = t/HP;
    int bt = (int)(btd >> 5), d = (int)(btd & 31);
    float ar = 0.f, ai = 0.f;
#pragma unroll
    for (int kh = 0; kh < 16; ++kh){
        float khf = (kh < 8) ? (float)kh : (float)(kh - 16);
        float s, c; __sincosf(2.f*PI_F*khf*(float)y/(float)HP, &s, &c);
        long mb = ((long)(kh*8 + kw)*NBT + bt)*64 + d;
        float re = Mm[mb], im = Mm[mb + 32];
        ar += c*re - s*im;
        ai += c*im + s*re;
    }
    T1[i*2] = ar; T1[i*2 + 1] = ai;
}

// ---- inverse DFT along x + residual + gelu (+optional skip save) ------------
__global__ void k_idft_x(const float* __restrict__ T1, float* __restrict__ z,
                         float* __restrict__ save){
    __shared__ float ct[NM2][WP], st[NM2][WP];
    for (int i = threadIdx.x; i < NM2*WP; i += blockDim.x){
        int kw = i/WP, xx = i % WP;
        float s, c; __sincosf(2.f*PI_F*(float)kw*(float)xx/(float)WP, &s, &c);
        ct[kw][xx] = c; st[kw][xx] = s;
    }
    __syncthreads();
    long row = (long)blockIdx.x*blockDim.x + threadIdx.x;  // btd*70 + y
    if (row >= (long)NBT*ND*HP) return;
    const float* tr = T1 + row*16;
    float re[NM2], im[NM2];
#pragma unroll
    for (int k = 0; k < NM2; ++k){
        float mlt = (k == 0) ? 1.f : 2.f;
        re[k] = tr[k*2]*mlt; im[k] = tr[k*2 + 1]*mlt;
    }
    float* zr = z + row*WP;
    float* sr = save ? save + row*WP : nullptr;
    for (int xx = 0; xx < WP; ++xx){
        float s = 0.f;
#pragma unroll
        for (int k = 0; k < NM2; ++k) s += re[k]*ct[k][xx] - im[k]*st[k][xx];
        float v = gelu_f(s*(1.f/(float)(HP*WP)) + zr[xx]);
        zr[xx] = v;
        if (sr) sr[xx] = v;
    }
}

// ---- proj MLP 32->128->1 on cropped interior --------------------------------
__global__ void k_proj(const float* __restrict__ z, const float* __restrict__ w1,
                       const float* __restrict__ b1, const float* __restrict__ w2,
                       const float* __restrict__ b2, float* __restrict__ out){
    __shared__ float sw1[NLIFT*ND];
    __shared__ float sw2[NLIFT];
    for (int i = threadIdx.x; i < NLIFT*ND; i += blockDim.x) sw1[i] = w1[i];
    for (int i = threadIdx.x; i < NLIFT; i += blockDim.x) sw2[i] = w2[i];
    __syncthreads();
    int idx = blockIdx.x*blockDim.x + threadIdx.x;
    if (idx >= NB*NTOK*HH*WW) return;
    int xx = idx & 63, yy = (idx >> 6) & 63; int bt = idx >> 12;
    float zc[ND];
    const float* zp = z + (long)bt*ND*PPIX + yy*WP + xx;
#pragma unroll
    for (int d = 0; d < ND; ++d) zc[d] = zp[(long)d*PPIX];
    float acc = b2[0];
    for (int f = 0; f < NLIFT; ++f){
        float hd = b1[f];
#pragma unroll
        for (int d = 0; d < ND; ++d) hd += sw1[f*ND + d]*zc[d];
        acc += sw2[f]*gelu_f(hd);
    }
    out[idx] = acc;
}

// ---- host launch ------------------------------------------------------------
extern "C" void kernel_launch(void* const* d_in, const int* in_sizes, int n_in,
                              void* d_out, int out_size, void* d_ws, size_t ws_size,
                              hipStream_t stream) {
    (void)in_sizes; (void)n_in; (void)out_size; (void)ws_size;
    // jax pytree (alphabetical-key) flatten order:
    // 0:x 1:fusion_b 2:fusion_w 3..18:layers[l]{k,mix,q,v} 19:lift_b1 20:lift_b2
    // 21:lift_w1 22:lift_w2 23:proj_b1 24:proj_b2 25:proj_w1 26:proj_w2
    const float* x        = (const float*)d_in[0];
    const float* fusion_b = (const float*)d_in[1];
    const float* fusion_w = (const float*)d_in[2];
    const float* lw[16];
    for (int l = 0; l < NLAY; ++l){
        const float* wk = (const float*)d_in[3 + l*4 + 0];
        const float* wm = (const float*)d_in[3 + l*4 + 1];
        const float* wq = (const float*)d_in[3 + l*4 + 2];
        const float* wv = (const float*)d_in[3 + l*4 + 3];
        lw[l*4 + 0] = wq; lw[l*4 + 1] = wk; lw[l*4 + 2] = wv; lw[l*4 + 3] = wm;
    }
    const float* lift_b1 = (const float*)d_in[19];
    const float* lift_b2 = (const float*)d_in[20];
    const float* lift_w1 = (const float*)d_in[21];
    const float* lift_w2 = (const float*)d_in[22];
    const float* proj_b1 = (const float*)d_in[23];
    const float* proj_b2 = (const float*)d_in[24];
    const float* proj_w1 = (const float*)d_in[25];
    const float* proj_w2 = (const float*)d_in[26];

    float* ws = (float*)d_ws;
    const long n_fused = (long)NB*NTOK*HH*WW;            // 655,360
    const long n_z     = (long)NBT*ND*PPIX;              // 25,088,000
    const long n_P     = (long)NBT*ND*HP*16;             // 5,734,400
    const long n_Zm    = (long)NBT*ND*NMODE*2;           // 1,310,720
    float* fused  = ws;
    float* zbuf   = fused + n_fused;
    float* out0   = zbuf + n_z;
    float* out1   = out0 + n_z;
    float* P      = out1 + n_z;
    float* T1     = P + n_P;
    float* Zm     = T1 + n_P;
    float* Qm     = Zm + n_Zm;
    float* Km     = Qm + n_Zm;
    float* Vm     = Km + n_Zm;
    float* Om     = Vm + n_Zm;
    float* Mm     = Om + n_Zm;
    float* scores = Mm + n_Zm;
    _Float16* hb      = (_Float16*)(scores + NB*NHEADS*NTOK*NTOK);
    _Float16* liftw2h = hb;                              // 4096 (packed frags)
    _Float16* ftab    = liftw2h + NLIFT*ND;              // 1536 (packed frags)
    _Float16* wpack   = ftab + 3*512;                    // 16 convs x 128*4096

    // weight prep (deterministic per call)
    k_pack_liftw2<<<16, 256, 0, stream>>>(lift_w2, liftw2h);
    k_pack_ftab<<<6, 256, 0, stream>>>(ftab);
    for (int cv = 0; cv < 16; ++cv)
        k_pack_spec<<<2048, 256, 0, stream>>>(lw[cv], wpack + (long)cv*NMODE*4096);

    // fusion conv + lift into padded z
    k_fusion<<<(int)((n_fused + 255)/256), 256, 0, stream>>>(x, fusion_w, fusion_b, fused);
    k_zero<<<4096, 256, 0, stream>>>(zbuf, n_z);
    k_lift<<<NBT*256, 32, 0, stream>>>(fused, lift_w1, lift_b1, liftw2h, lift_b2, zbuf);

    dim3 gmix(NMODE, NBT/16);
    for (int l = 0; l < NLAY; ++l){
        if (l == 2) k_add<<<4096, 256, 0, stream>>>(zbuf, out1, n_z);
        if (l == 3) k_add<<<4096, 256, 0, stream>>>(zbuf, out0, n_z);
        // forward truncated DFT of z -> Zm (modes)
        k_dft_x<<<(int)(NBT*ND*HP/16), 32, 0, stream>>>(zbuf, ftab, P);
        k_dft_y<<<(int)((NBT*ND*NMODE + 255)/256), 256, 0, stream>>>(P, Zm);
        // q,k,v channel mixes in mode space
        k_modemix<<<gmix, 32, 0, stream>>>(Zm, wpack + (long)(l*4 + 0)*NMODE*4096, Qm);
        k_modemix<<<gmix, 32, 0, stream>>>(Zm, wpack + (long)(l*4 + 1)*NMODE*4096, Km);
        k_modemix<<<gmix, 32, 0, stream>>>(Zm, wpack + (long)(l*4 + 2)*NMODE*4096, Vm);
        // attention (Parseval in mode space)
        k_scores<<<NB*NHEADS*NTOK*NTOK, 256, 0, stream>>>(Qm, Km, scores);
        k_softmax<<<1, 320, 0, stream>>>(scores);
        k_attn_apply<<<(int)((NBT*ND*NMODE + 255)/256), 256, 0, stream>>>(scores, Vm, Om);
        // mix spectral conv on o (band-limited -> pure mode-space channel mix)
        k_modemix<<<gmix, 32, 0, stream>>>(Om, wpack + (long)(l*4 + 3)*NMODE*4096, Mm);
        // inverse DFT + residual + gelu (save skip outputs for layers 0/1)
        k_idft_y<<<(int)((NBT*ND*HP*NM2 + 255)/256), 256, 0, stream>>>(Mm, T1);
        float* save = (l == 0) ? out0 : (l == 1) ? out1 : nullptr;
        k_idft_x<<<(int)((NBT*ND*HP + 127)/128), 128, 0, stream>>>(T1, zbuf, save);
    }
    // crop + proj MLP
    k_proj<<<(int)((n_fused + 255)/256), 256, 0, stream>>>(zbuf, proj_w1, proj_b1,
                                                           proj_w2, proj_b2, (float*)d_out);
}
